// GaussProjDrop_37812892074180
// MI455X (gfx1250) — matrix operand order repaired and checked by reading, not compiled, not run on hardware
//
#include <hip/hip_runtime.h>
#include <stdint.h>

// GaussProjDrop on MI455X (gfx1250):
//   out[r, ans[r,j]] = (1/3) * sum_k x[r, ans[r,k]] * U[k,j] + x[r, ans[r,j]]
//   ans[r,j] = (perm[j] * s_r) & 1023,  s_r = (6*randint*r + 1) & 1023 (odd -> permutation)
// Fused gather+GEMM+scatter, bf16 hi/lo split (3x v_wmma_f32_16x16x32_bf16),
// double-buffered LDS pipeline: one barrier/iter, next-tile staging overlaps WMMA.

typedef __attribute__((ext_vector_type(16))) __bf16    v16bf;
typedef __attribute__((ext_vector_type(8)))  float     v8f;
typedef __attribute__((ext_vector_type(4)))  uint32_t  uint4v;

#define C_DIM   1024
#define M_ROWS  16384      // B*S = 8*2048
#define BM      128
#define BN      128
#define BK      32
#define KSTEPS  (C_DIM / BK)
#define LDA     20         // dword stride of A LDS tile rows (16 dw data + 4 pad)
#define LDB     20

union FragBF { uint32_t u[8]; uint4v q[2]; v16bf bf; };
union AccF   { float    f[8]; v8f   v;  };

__device__ __forceinline__ uint32_t f2bf_rne(float f) {
    uint32_t u = __float_as_uint(f);
    return (u + 0x7FFFu + ((u >> 16) & 1u)) >> 16;
}

// ---- prep: split U into bf16 hi/lo, stored TRANSPOSED (Ut[n][k])
__global__ __launch_bounds__(256) void u_split_kernel(
    const float* __restrict__ U, uint16_t* __restrict__ Ut_hi, uint16_t* __restrict__ Ut_lo)
{
    int idx = blockIdx.x * 256 + threadIdx.x;     // idx = k*1024 + n (coalesced read)
    int k = idx >> 10;
    int n = idx & (C_DIM - 1);
    float v = U[idx];
    uint32_t hb = f2bf_rne(v);
    uint32_t lb = f2bf_rne(v - __uint_as_float(hb << 16));
    Ut_hi[(size_t)n * C_DIM + k] = (uint16_t)hb;
    Ut_lo[(size_t)n * C_DIM + k] = (uint16_t)lb;
}

// staging registers for one k-tile (per thread: 16 A floats, 8+8 B dwords)
struct Stage {
    float    av[16];
    uint32_t bh[8], bl[8];
};

__device__ __forceinline__ void stage_issue(
    Stage& st, const float* __restrict__ x,
    const uint32_t* __restrict__ BgH, const uint32_t* __restrict__ BgL,
    const uint16_t* permS, int t, int block_m, int block_n, uint32_t mul, int k0)
{
#pragma unroll
    for (int p = 0; p < 8; ++p) {               // B tile: 128 n-rows x 16 dwords
        int idx = t + p * 256;
        int n = idx >> 4, ku = idx & 15;
        size_t g = (size_t)(block_n + n) * (C_DIM >> 1) + (k0 >> 1) + ku;
        st.bh[p] = BgH[g];
        st.bl[p] = BgL[g];
    }
#pragma unroll
    for (int p = 0; p < 8; ++p) {               // A tile: per-row permuted gather (L2-resident x)
        int idx = t + p * 256;
        int rr = idx >> 4, kp = idx & 15;
        int R = block_m + rr;
        uint32_t s = (mul * (uint32_t)R + 1u) & 1023u;
        int ka = k0 + kp * 2;
        uint32_t c0 = ((uint32_t)permS[ka]     * s) & 1023u;
        uint32_t c1 = ((uint32_t)permS[ka + 1] * s) & 1023u;
        st.av[2 * p]     = x[(size_t)R * C_DIM + c0];
        st.av[2 * p + 1] = x[(size_t)R * C_DIM + c1];
    }
}

__device__ __forceinline__ void stage_commit(
    const Stage& st, uint32_t* AsH, uint32_t* AsL, uint32_t* BsH, uint32_t* BsL, int t)
{
#pragma unroll
    for (int p = 0; p < 8; ++p) {
        int idx = t + p * 256;
        int n = idx >> 4, ku = idx & 15;
        BsH[n * LDB + ku] = st.bh[p];
        BsL[n * LDB + ku] = st.bl[p];
    }
#pragma unroll
    for (int p = 0; p < 8; ++p) {
        int idx = t + p * 256;
        int rr = idx >> 4, kp = idx & 15;
        float v0 = st.av[2 * p], v1 = st.av[2 * p + 1];
        uint32_t h0 = f2bf_rne(v0), h1 = f2bf_rne(v1);
        uint32_t l0 = f2bf_rne(v0 - __uint_as_float(h0 << 16));
        uint32_t l1 = f2bf_rne(v1 - __uint_as_float(h1 << 16));
        AsH[rr * LDA + kp] = h0 | (h1 << 16);
        AsL[rr * LDA + kp] = l0 | (l1 << 16);
    }
}

__global__ __launch_bounds__(256) void gauss_proj_gemm(
    const float*    __restrict__ x,
    const uint16_t* __restrict__ Ut_hi,
    const uint16_t* __restrict__ Ut_lo,
    const int*      __restrict__ perm,
    const int*      __restrict__ randint_p,
    float*          __restrict__ out)
{
    __shared__ __align__(16) uint32_t AsH[2][BM * LDA];
    __shared__ __align__(16) uint32_t AsL[2][BM * LDA];
    __shared__ __align__(16) uint32_t BsH[2][BN * LDB];
    __shared__ __align__(16) uint32_t BsL[2][BN * LDB];
    __shared__ uint16_t permS[C_DIM];

    const int t       = threadIdx.x;
    const int block_n = blockIdx.x * BN;
    const int block_m = blockIdx.y * BM;
    const uint32_t mul = (uint32_t)(6 * randint_p[0]);   // mod-2^10: wraparound-safe

    for (int i = t; i < C_DIM; i += 256) permS[i] = (uint16_t)perm[i];
    __syncthreads();

    const int lane  = t & 31;
    const int wave  = t >> 5;
    const int wm    = wave & 3;      // 4 waves along M (32 rows each)
    const int wn    = wave >> 2;     // 2 waves along N (64 cols each)
    const int lrow  = lane & 15;
    const int lhalf = lane >> 4;

    AccF acc[2][4];
#pragma unroll
    for (int mt = 0; mt < 2; ++mt)
#pragma unroll
        for (int nt = 0; nt < 4; ++nt)
#pragma unroll
            for (int i = 0; i < 8; ++i) acc[mt][nt].f[i] = 0.0f;

    const uint32_t* BgH = (const uint32_t*)Ut_hi;   // dword view (2 bf16 per dword)
    const uint32_t* BgL = (const uint32_t*)Ut_lo;

    // prologue: fill buffer 0
    Stage st;
    stage_issue(st, x, BgH, BgL, permS, t, block_m, block_n, mul, 0);
    stage_commit(st, AsH[0], AsL[0], BsH[0], BsL[0], t);
    __syncthreads();

    const int a_base = (wm * 32 + lrow) * LDA + lhalf * 4;   // + mt*16*LDA
    const int b_base = (wn * 64 + lrow) * LDB + lhalf * 4;   // + nt*16*LDB

    for (int kt = 0; kt < KSTEPS; ++kt) {
        const int  cur      = kt & 1;
        const bool has_next = (kt + 1 < KSTEPS);

        // issue next tile's global loads early (latency overlapped with WMMA below)
        if (has_next)
            stage_issue(st, x, BgH, BgL, permS, t, block_m, block_n, mul, (kt + 1) * BK);

        // fragment loads from current buffer: clean ds_load_b128 pairs
        FragBF aH[2], aL[2], bH[4], bL[4];
#pragma unroll
        for (int mt = 0; mt < 2; ++mt) {
            int base = a_base + mt * 16 * LDA;
            aH[mt].q[0] = *(const uint4v*)&AsH[cur][base];
            aH[mt].q[1] = *(const uint4v*)&AsH[cur][base + 8];
            aL[mt].q[0] = *(const uint4v*)&AsL[cur][base];
            aL[mt].q[1] = *(const uint4v*)&AsL[cur][base + 8];
        }
#pragma unroll
        for (int nt = 0; nt < 4; ++nt) {
            int base = b_base + nt * 16 * LDB;
            bH[nt].q[0] = *(const uint4v*)&BsH[cur][base];
            bH[nt].q[1] = *(const uint4v*)&BsH[cur][base + 8];
            bL[nt].q[0] = *(const uint4v*)&BsL[cur][base];
            bL[nt].q[1] = *(const uint4v*)&BsL[cur][base + 8];
        }

        // commit next tile into the idle buffer (VALU split co-executes with XDL WMMA)
        if (has_next)
            stage_commit(st, AsH[cur ^ 1], AsL[cur ^ 1], BsH[cur ^ 1], BsL[cur ^ 1], t);

        // 3-product bf16 split: aH*bH + aH*bL + aL*bH  (fp32 accumulate)
#pragma unroll
        for (int mt = 0; mt < 2; ++mt)
#pragma unroll
            for (int nt = 0; nt < 4; ++nt) {
                acc[mt][nt].v = __builtin_amdgcn_wmma_f32_16x16x32_bf16(
                    false, aH[mt].bf, false, bH[nt].bf, (short)0, acc[mt][nt].v, false, false);
                acc[mt][nt].v = __builtin_amdgcn_wmma_f32_16x16x32_bf16(
                    false, aH[mt].bf, false, bL[nt].bf, (short)0, acc[mt][nt].v, false, false);
                acc[mt][nt].v = __builtin_amdgcn_wmma_f32_16x16x32_bf16(
                    false, aL[mt].bf, false, bH[nt].bf, (short)0, acc[mt][nt].v, false, false);
            }
        __syncthreads();   // single barrier per iteration (double-buffered)
    }

    // epilogue: scatter through ans + residual.  C/D layout:
    //   VGPR i: lanes 0-15 -> M=i, N=lane; lanes 16-31 -> M=8+i, N=lane-16
    const float scale = 1.0f / 3.0f;
#pragma unroll
    for (int mt = 0; mt < 2; ++mt)
#pragma unroll
        for (int nt = 0; nt < 4; ++nt) {
            int jcol = block_n + wn * 64 + nt * 16 + lrow;
            uint32_t pj = permS[jcol];
#pragma unroll
            for (int i = 0; i < 8; ++i) {
                int R = block_m + wm * 32 + mt * 16 + lhalf * 8 + i;
                uint32_t s = (mul * (uint32_t)R + 1u) & 1023u;
                uint32_t a = (pj * s) & 1023u;
                size_t o = (size_t)R * C_DIM + a;
                out[o] = acc[mt][nt].f[i] * scale + x[o];
            }
        }
}

extern "C" void kernel_launch(void* const* d_in, const int* in_sizes, int n_in,
                              void* d_out, int out_size, void* d_ws, size_t ws_size,
                              hipStream_t stream) {
    const float* x       = (const float*)d_in[0];
    const float* U       = (const float*)d_in[1];
    const int*   perm    = (const int*)d_in[2];
    const int*   randint = (const int*)d_in[3];
    float*       out     = (float*)d_out;

    uint16_t* Ut_hi = (uint16_t*)d_ws;                           // 2 MB
    uint16_t* Ut_lo = Ut_hi + (size_t)C_DIM * C_DIM;             // 2 MB

    u_split_kernel<<<(C_DIM * C_DIM) / 256, 256, 0, stream>>>(U, Ut_hi, Ut_lo);

    dim3 grid(C_DIM / BN, M_ROWS / BM);   // (8, 128)
    gauss_proj_gemm<<<grid, 256, 0, stream>>>(x, Ut_hi, Ut_lo, perm, randint, out);
}
